// ConditionalPhysicalForward_75935021794152
// MI455X (gfx1250) — compile-verified, matrix operand order checked by reading, not run on hardware
//
#include <hip/hip_runtime.h>
#include <hip/hip_bf16.h>

// ---------------------------------------------------------------------------
// MI455X (gfx1250) implementation.
// Conv layers -> implicit GEMM via v_wmma_f32_16x16x32_f16 (wave32), with
//   block-shared im2col B-fragments in LDS and (layers 1-2) TDM-staged
//   weight slabs (tensor_load_to_lds + s_wait_tensorcnt).
// fc1 -> WMMA GEMM (16x128x8192) on pre-packed fragments.
// GroupNorm -> two-pass (atomic stats + fused norm/relu/maxpool).
// NFW field -> elementwise VALU kernel.
// ---------------------------------------------------------------------------

typedef __attribute__((ext_vector_type(16))) _Float16 v16h;
typedef __attribute__((ext_vector_type(8)))  float    v8f;
typedef __attribute__((ext_vector_type(4)))  unsigned int v4u;
typedef __attribute__((ext_vector_type(8)))  int v8i_;
typedef __attribute__((ext_vector_type(4)))  int v4i_;

// ISA 7.12.2: 16-bit 16x32 fragment. For lane L, element e (0..15):
//   vgpr v = e>>1, pair p = e&1; lanes 16-31 add K base 8;
//   v<4 -> K = base + 2v + p ; v>=4 -> K = base + 16 + 2(v-4) + p
__device__ __forceinline__ int frag_k(int lane, int e) {
  int v = e >> 1, p = e & 1;
  int base = (lane & 16) ? 8 : 0;
  return (v < 4) ? (base + 2 * v + p) : (base + 16 + 2 * (v - 4) + p);
}

// --------------------------- weight pre-pack (A: MxK) ---------------------
// idx = ((coTile*nK + kstep)*32 + lane)*16 + e
__global__ void pack_w_kernel(const float* __restrict__ w, _Float16* __restrict__ dst,
                              int Cout, int K, int nK) {
  int i = blockIdx.x * 256 + threadIdx.x;
  int total = (Cout / 16) * nK * 512;
  if (i >= total) return;
  int e = i & 15;
  int lane = (i >> 4) & 31;
  int rest = i >> 9;
  int kstep = rest % nK;
  int coTile = rest / nK;
  int m = lane & 15;
  int k = kstep * 32 + frag_k(lane, e);
  float v = 0.f;
  if (k < K) v = w[(coTile * 16 + m) * K + k];
  dst[i] = (_Float16)v;
}

// ------------------- generic B-matrix (KxN, row-major) pre-pack -----------
__global__ void pack_bmat_kernel(const float* __restrict__ src, _Float16* __restrict__ dst,
                                 int K, int N, int nK) {
  int i = blockIdx.x * 256 + threadIdx.x;
  int total = (N / 16) * nK * 512;
  if (i >= total) return;
  int e = i & 15;
  int lane = (i >> 4) & 31;
  int rest = i >> 9;
  int kstep = rest % nK;
  int nTile = rest / nK;
  int n = nTile * 16 + (lane & 15);
  int k = kstep * 32 + frag_k(lane, e);
  float v = 0.f;
  if (k < K) v = src[(size_t)k * N + n];
  dst[i] = (_Float16)v;
}

// -------------------- pack feats (16x8192 fp32) as A-fragments ------------
__global__ void pack_feats_kernel(const float* __restrict__ feats, _Float16* __restrict__ dst) {
  int i = blockIdx.x * 256 + threadIdx.x;  // 256 ksteps * 512 = 131072
  if (i >= 131072) return;
  int e = i & 15;
  int lane = (i >> 4) & 31;
  int kstep = i >> 9;
  int m = lane & 15;
  int k = kstep * 32 + frag_k(lane, e);
  dst[i] = (_Float16)feats[m * 8192 + k];
}

// --------------------------- fp32 -> f16 convert --------------------------
__global__ void cvt_f16_kernel(const float* __restrict__ src, _Float16* __restrict__ dst, int n) {
  int i = blockIdx.x * 256 + threadIdx.x;
  if (i < n) dst[i] = (_Float16)src[i];
}

// --------------------------- conv (implicit GEMM, WMMA) -------------------
// One block = 16-pixel strip (row y, x0..x0+15) of one image.
// blockDim = (Cout/16)*32; wave w -> output channels [16w,16w+16).
// LDS: halo patch [Cin][3][18] -> shared B-fragments [NK][32][16]
//      (+ TDM-staged weight slab for USE_TDM layers).
template <int NK, bool USE_TDM>
__global__ void conv_wmma_kernel(const _Float16* __restrict__ in,   // [Cin][H][W]
                                 const _Float16* __restrict__ wpk,  // packed A frags
                                 const float* __restrict__ bias,    // [Cout]
                                 float* __restrict__ out,           // [Cout][H][W]
                                 int Cin, int H, int W, int logW16) {
  extern __shared__ _Float16 lds[];
  const int y = blockIdx.x >> logW16;
  const int x0 = (blockIdx.x & ((1 << logW16) - 1)) << 4;
  const int wave = threadIdx.x >> 5;
  const int lane = threadIdx.x & 31;
  const int K = Cin * 9;
  const int fragOff = (Cin * 54 + 15) & ~15;
  _Float16* patch = lds;
  _Float16* fragB = lds + fragOff;
  _Float16* wlds = fragB + NK * 512;

  if (USE_TDM) {
    // TDM: stage the whole packed weight slab for this block into LDS.
    // D# per ISA 8.3/8.4: 1-D tile, count=1, data_size=2B, type=2.
    if (threadIdx.x < 32) {  // wave 0 issues; EXEC ignored by TENSOR ops
      unsigned long long ga = (unsigned long long)(size_t)wpk;
      unsigned int ldsOffB = (unsigned)((fragOff + NK * 512) * sizeof(_Float16));
      unsigned int elems = (unsigned)((blockDim.x >> 5) * NK * 512);
      v4u g0 = {1u, ldsOffB, (unsigned)ga,
                (unsigned)((ga >> 32) & 0x01FFFFFFull) | 0x80000000u};
      v8i_ g1 = {0x10000,                              // data_size = 2B
                 (int)((elems & 0xFFFFu) << 16),       // tensor_dim0 [15:0]
                 (int)((elems >> 16) & 0xFFFFu),       // tensor_dim0 [31:16]
                 (int)((elems & 0xFFFFu) << 16),       // tile_dim0
                 0,
                 (int)elems,                           // tensor_dim0_stride lo
                 0, 0};
      v4i_ gz = {0, 0, 0, 0};
#if __clang_major__ >= 23
      v8i_ gz8 = {0, 0, 0, 0, 0, 0, 0, 0};
      __builtin_amdgcn_tensor_load_to_lds(g0, g1, gz, gz, gz8, 0);
#else
      __builtin_amdgcn_tensor_load_to_lds(g0, g1, gz, gz, 0);
#endif
      __builtin_amdgcn_s_wait_tensorcnt(0);
    }
  }

  // Stage halo patch [ci][ry 0..2][rx 0..17] with zero borders.
  const int total = Cin * 54;
  for (int i = threadIdx.x; i < total; i += blockDim.x) {
    int ci = i / 54;
    int r = i - ci * 54;
    int ry = r / 18;
    int rx = r - ry * 18;
    int iy = y + ry - 1;
    int ix = x0 + rx - 1;
    _Float16 v = (_Float16)0.f;
    if (iy >= 0 && iy < H && ix >= 0 && ix < W) v = in[(ci * H + iy) * W + ix];
    patch[i] = v;
  }
  __syncthreads();

  // Build shared im2col B-fragments once per block (fragment-exact order).
  for (int i = threadIdx.x; i < NK * 512; i += blockDim.x) {
    int e = i & 15;
    int ln = (i >> 4) & 31;
    int kstep = i >> 9;
    int n2 = ln & 15;
    int k = kstep * 32 + frag_k(ln, e);
    _Float16 v = (_Float16)0.f;
    if (k < K) {
      int ci = k / 9;
      int r = k - ci * 9;
      int ky = r / 3;
      int kx = r - ky * 3;
      v = patch[ci * 54 + ky * 18 + n2 + kx];
    }
    fragB[i] = v;
  }
  __syncthreads();

  v8f c = {0.f, 0.f, 0.f, 0.f, 0.f, 0.f, 0.f, 0.f};
  const _Float16* wbase = USE_TDM ? (wlds + (size_t)wave * NK * 512)
                                  : (wpk + (size_t)wave * NK * 512);
#pragma unroll
  for (int kstep = 0; kstep < NK; ++kstep) {
    const _Float16* ap = wbase + ((size_t)kstep * 32 + lane) * 16;
    v16h a = *(const v16h*)ap;  // contiguous 32B (global or LDS)
    if (!USE_TDM && kstep + 1 < NK)
      __builtin_prefetch(wbase + ((size_t)(kstep + 1) * 32 + lane) * 16, 0, 1);
    v16h b = *(const v16h*)(fragB + ((size_t)kstep * 32 + lane) * 16);  // 2x ds_load_b128
    c = __builtin_amdgcn_wmma_f32_16x16x32_f16(false, a, false, b,
                                               (short)0, c, false, false);
  }

  // D layout: lane -> N = lane&15 ; vgpr r -> M = r (+8 for lanes 16-31).
  const int n = lane & 15;
  const int m0 = wave * 16 + ((lane & 16) ? 8 : 0);
#pragma unroll
  for (int r = 0; r < 8; ++r) {
    int co = m0 + r;
    out[((size_t)co * H + y) * W + (x0 + n)] = c[r] + bias[co];
  }
}

// --------------------------- GroupNorm stats ------------------------------
__global__ void zero_kernel(float* p, int n) {
  int i = blockIdx.x * 64 + threadIdx.x;
  if (i < n) p[i] = 0.f;
}

__global__ void gn_stats_kernel(const float* __restrict__ conv, float* __restrict__ stats,
                                int chPerGroup, int HW, int blocksPerGroup) {
  __shared__ float rs0[256], rs1[256];
  int g = blockIdx.x / blocksPerGroup;
  int bi = blockIdx.x % blocksPerGroup;
  size_t base = (size_t)g * chPerGroup * HW;
  size_t tot = (size_t)chPerGroup * HW;
  float s = 0.f, ss = 0.f;
  for (size_t i = (size_t)bi * blockDim.x + threadIdx.x; i < tot;
       i += (size_t)blocksPerGroup * blockDim.x) {
    float v = conv[base + i];
    s += v;
    ss += v * v;
  }
  rs0[threadIdx.x] = s;
  rs1[threadIdx.x] = ss;
  __syncthreads();
  for (int off = 128; off > 0; off >>= 1) {
    if ((int)threadIdx.x < off) {
      rs0[threadIdx.x] += rs0[threadIdx.x + off];
      rs1[threadIdx.x] += rs1[threadIdx.x + off];
    }
    __syncthreads();
  }
  if (threadIdx.x == 0) {
    atomicAdd(&stats[g * 2 + 0], rs0[0]);
    atomicAdd(&stats[g * 2 + 1], rs1[0]);
  }
}

// ----------------- GroupNorm apply + ReLU + 2x2 maxpool -> f16 ------------
__global__ void gn_pool_kernel(const float* __restrict__ conv, const float* __restrict__ stats,
                               const float* __restrict__ gamma, const float* __restrict__ beta,
                               _Float16* __restrict__ outp, int C, int H, int W,
                               int chPerGroup, float invN) {
  int Ho = H >> 1, Wo = W >> 1;
  int i = blockIdx.x * 256 + threadIdx.x;
  int total = C * Ho * Wo;
  if (i >= total) return;
  int c = i / (Ho * Wo);
  int rem = i - c * (Ho * Wo);
  int oy = rem / Wo, ox = rem - (rem / Wo) * Wo;
  int g = c / chPerGroup;
  float mean = stats[g * 2] * invN;
  float var = stats[g * 2 + 1] * invN - mean * mean;
  float rstd = rsqrtf(var + 1e-5f);
  float ga = gamma[c], be = beta[c];
  const float* p = conv + ((size_t)c * H + oy * 2) * W + ox * 2;
  float m = 0.f;  // relu then max == max(0, ...)
  m = fmaxf(m, (p[0] - mean) * rstd * ga + be);
  m = fmaxf(m, (p[1] - mean) * rstd * ga + be);
  m = fmaxf(m, (p[W] - mean) * rstd * ga + be);
  m = fmaxf(m, (p[W + 1] - mean) * rstd * ga + be);
  outp[i] = (_Float16)m;
}

// ----------------- 4x4 avgpool (32x32 -> 8x8) -> features -----------------
__global__ void avgpool4_kernel(const _Float16* __restrict__ inp, float* __restrict__ feats) {
  int i = blockIdx.x * 256 + threadIdx.x;  // 128*8*8 = 8192
  if (i >= 8192) return;
  int c = i >> 6;
  int oy = (i >> 3) & 7;
  int ox = i & 7;
  const _Float16* p = inp + ((size_t)c * 32 + oy * 4) * 32 + ox * 4;
  float s = 0.f;
#pragma unroll
  for (int ky = 0; ky < 4; ++ky)
#pragma unroll
    for (int kx = 0; kx < 4; ++kx) s += (float)p[ky * 32 + kx];
  feats[i] = s * (1.f / 16.f);
}

// ----------------- fc1 via WMMA: (16 x 8192) @ (8192 x 128) ---------------
// One block, 8 waves; wave j -> output columns [16j,16j+16). nK = 256.
__global__ void fc1_wmma_kernel(const _Float16* __restrict__ apk,
                                const _Float16* __restrict__ bpk,
                                const float* __restrict__ b1, float* __restrict__ h1) {
  int wave = threadIdx.x >> 5;
  int lane = threadIdx.x & 31;
  v8f c = {0.f, 0.f, 0.f, 0.f, 0.f, 0.f, 0.f, 0.f};
  const _Float16* bb = bpk + (size_t)wave * 256 * 512;
  for (int kstep = 0; kstep < 256; ++kstep) {
    v16h a = *(const v16h*)(apk + ((size_t)kstep * 32 + lane) * 16);
    v16h b = *(const v16h*)(bb + ((size_t)kstep * 32 + lane) * 16);
    c = __builtin_amdgcn_wmma_f32_16x16x32_f16(false, a, false, b,
                                               (short)0, c, false, false);
  }
  int j = wave * 16 + (lane & 15);
  int m0 = (lane & 16) ? 8 : 0;
#pragma unroll
  for (int r = 0; r < 8; ++r) {
    int m = m0 + r;  // batch index
    h1[m * 128 + j] = fmaxf(c[r] + b1[j], 0.f);
  }
}

// --------------------- fc2 + four scalar heads per batch ------------------
__global__ void fc2_heads_kernel(const float* __restrict__ h1, const float* __restrict__ w2,
                                 const float* __restrict__ b2,
                                 const float* ksw, const float* ksb,
                                 const float* rsw, const float* rsb,
                                 const float* x0w, const float* x0b,
                                 const float* y0w, const float* y0b,
                                 float* __restrict__ scalars) {
  __shared__ float h2[64];
  int b = blockIdx.x;
  int j = threadIdx.x;  // 64 threads
  float s = b2[j];
  for (int i = 0; i < 128; ++i) s += h1[b * 128 + i] * w2[i * 64 + j];
  h2[j] = fmaxf(s, 0.f);
  __syncthreads();
  if (j < 4) {
    const float* w = (j == 0) ? ksw : (j == 1) ? rsw : (j == 2) ? x0w : y0w;
    float bb = (j == 0) ? ksb[0] : (j == 1) ? rsb[0] : (j == 2) ? x0b[0] : y0b[0];
    float t = bb;
    for (int i = 0; i < 64; ++i) t += h2[i] * w[i];
    float r;
    if (j == 0)      r = 0.005f + 0.495f / (1.f + expf(-t));
    else if (j == 1) r = 0.05f + 0.45f / (1.f + expf(-t));
    else             r = 0.2f * tanhf(t);
    scalars[b * 4 + j] = r;
  }
}

// --------------------------- NFW field kernel -----------------------------
__global__ void nfw_kernel(const float* __restrict__ scalars, float* __restrict__ out,
                           int Bt, int H, int W) {
  size_t i = (size_t)blockIdx.x * 256 + threadIdx.x;
  size_t total = (size_t)Bt * H * W;
  if (i >= total) return;
  int x = (int)(i % W);
  int y = (int)((i / W) % H);
  int b = (int)(i / ((size_t)W * H));
  float ks = scalars[b * 4 + 0];
  float rs = scalars[b * 4 + 1];
  float x0 = scalars[b * 4 + 2];
  float y0 = scalars[b * 4 + 3];
  float X = ((float)x - (W - 1) * 0.5f) * 0.05f;
  float Y = ((float)y - (H - 1) * 0.5f) * 0.05f;
  float dx = X - x0, dy = Y - y0;
  float r = sqrtf(dx * dx + dy * dy + 1e-8f);
  float xn = fminf(fmaxf(r / (rs + 1e-8f), 1e-6f), 1e6f);
  bool lt = xn < 1.f, gt = xn > 1.f;
  float xn2 = xn * xn;
  float sqrt_lt = sqrtf(fmaxf(1.f - xn2, 1e-12f));
  float u = sqrtf(fminf(fmaxf((1.f - xn) / (1.f + xn), 0.f), 1.f - 1e-6f));
  float atanh_u = 0.5f * (log1pf(u) - log1pf(-u));
  float a_lt = 4.f / (xn + 1e-12f) * (1.f - 2.f * atanh_u / (sqrt_lt + 1e-12f));
  float sqrt_gt = sqrtf(fmaxf(xn2 - 1.f, 1e-12f));
  float v = sqrtf(fmaxf((xn - 1.f) / (1.f + xn), 0.f));
  float atan_v = atanf(v);
  float a_gt = 4.f / (xn + 1e-12f) * (1.f - 2.f * atan_v / (sqrt_gt + 1e-12f));
  float alpha_r = (lt ? a_lt : (gt ? a_gt : 2.f)) * ks * rs;
  float ax = alpha_r * dx / (r + 1e-8f);
  float ay = alpha_r * dy / (r + 1e-8f);
  float denom = xn2 - 1.f;
  float k_lt = 2.f * ks / (denom + 1e-12f) * (1.f - 2.f / (sqrt_lt + 1e-12f) * atanh_u);
  float k_gt = 2.f * ks / (denom + 1e-12f) * (1.f - 2.f / (sqrt_gt + 1e-12f) * atan_v);
  float kap = lt ? k_lt : (gt ? k_gt : (2.f * ks / 3.f));
  size_t plane = (size_t)H * W;
  size_t base = (size_t)b * 3 * plane + (size_t)y * W + x;
  out[base] = ax;
  out[base + plane] = ay;
  out[base + 2 * plane] = kap;
}

// ===========================================================================
extern "C" void kernel_launch(void* const* d_in, const int* in_sizes, int n_in,
                              void* d_out, int out_size, void* d_ws, size_t ws_size,
                              hipStream_t stream) {
  (void)in_sizes; (void)n_in; (void)out_size; (void)ws_size;
  const float* x      = (const float*)d_in[0];
  const float* cw[4]  = {(const float*)d_in[1], (const float*)d_in[5],
                         (const float*)d_in[9], (const float*)d_in[13]};
  const float* cb[4]  = {(const float*)d_in[2], (const float*)d_in[6],
                         (const float*)d_in[10], (const float*)d_in[14]};
  const float* gng[4] = {(const float*)d_in[3], (const float*)d_in[7],
                         (const float*)d_in[11], (const float*)d_in[15]};
  const float* gnb[4] = {(const float*)d_in[4], (const float*)d_in[8],
                         (const float*)d_in[12], (const float*)d_in[16]};
  const float* w1 = (const float*)d_in[17];
  const float* b1 = (const float*)d_in[18];
  const float* w2 = (const float*)d_in[19];
  const float* b2 = (const float*)d_in[20];
  const float* ksw = (const float*)d_in[21]; const float* ksb = (const float*)d_in[22];
  const float* rsw = (const float*)d_in[23]; const float* rsb = (const float*)d_in[24];
  const float* x0w = (const float*)d_in[25]; const float* x0b = (const float*)d_in[26];
  const float* y0w = (const float*)d_in[27]; const float* y0b = (const float*)d_in[28];
  float* out = (float*)d_out;

  const int Cin[4]  = {1, 32, 64, 96};
  const int Cout[4] = {32, 64, 96, 128};
  const int Hs[4]   = {512, 256, 128, 64};
  const int logW16s[4] = {5, 4, 3, 2};
  int nK[4], Kk[4];
  for (int l = 0; l < 4; ++l) { Kk[l] = Cin[l] * 9; nK[l] = (Kk[l] + 31) / 32; }

  // Workspace layout (256B aligned); total ~50 MB.
  char* ws = (char*)d_ws;
  size_t off = 0;
  auto alloc = [&](size_t bytes) -> char* {
    off = (off + 255) & ~(size_t)255;
    char* p = ws + off;
    off += bytes;
    return p;
  };
  _Float16* wpk[4];
  for (int l = 0; l < 4; ++l)
    wpk[l] = (_Float16*)alloc((size_t)(Cout[l] / 16) * nK[l] * 512 * sizeof(_Float16));
  _Float16* in16   = (_Float16*)alloc((size_t)16 * 512 * 512 * sizeof(_Float16));
  float*    convbf = (float*)alloc((size_t)32 * 512 * 512 * sizeof(float));
  _Float16* poolA  = (_Float16*)alloc((size_t)32 * 256 * 256 * sizeof(_Float16));
  _Float16* poolB  = (_Float16*)alloc((size_t)64 * 128 * 128 * sizeof(_Float16));
  float*    stats  = (float*)alloc(16 * sizeof(float));
  float*    feats  = (float*)alloc((size_t)16 * 8192 * sizeof(float));
  float*    h1     = (float*)alloc((size_t)16 * 128 * sizeof(float));
  float*    scalars= (float*)alloc(64 * sizeof(float));
  _Float16* w1pk   = (_Float16*)alloc((size_t)8 * 256 * 512 * sizeof(_Float16));
  _Float16* featpk = (_Float16*)alloc((size_t)256 * 512 * sizeof(_Float16));

  // Pre-pack conv + fc1 weights into WMMA fragment order (f16).
  for (int l = 0; l < 4; ++l) {
    int tot = (Cout[l] / 16) * nK[l] * 512;
    pack_w_kernel<<<(tot + 255) / 256, 256, 0, stream>>>(cw[l], wpk[l], Cout[l], Kk[l], nK[l]);
  }
  {
    int tot = 8 * 256 * 512;
    pack_bmat_kernel<<<(tot + 255) / 256, 256, 0, stream>>>(w1, w1pk, 8192, 128, 256);
  }
  {
    int n = 16 * 512 * 512;
    cvt_f16_kernel<<<(n + 255) / 256, 256, 0, stream>>>(x, in16, n);
  }

  // Encoder, one batch element at a time (peak scratch ~50 MB).
  for (int b = 0; b < 16; ++b) {
    const _Float16* lin = in16 + (size_t)b * 512 * 512;
    _Float16* lpool[4] = {poolA, poolB, poolA, poolB};
    for (int l = 0; l < 4; ++l) {
      int H = Hs[l], W = Hs[l], Co = Cout[l], Ci = Cin[l];
      int threads = (Co / 16) * 32;
      int grid = H * (W / 16);
      int fragOff = (Ci * 54 + 15) & ~15;
      size_t ldsHalfs = (size_t)fragOff + (size_t)nK[l] * 512;
      if (l < 2) ldsHalfs += (size_t)(threads >> 5) * nK[l] * 512;  // TDM slab
      size_t ldsB = ldsHalfs * sizeof(_Float16);
      switch (l) {
        case 0: conv_wmma_kernel<1, true><<<grid, threads, ldsB, stream>>>(
                    lin, wpk[l], cb[l], convbf, Ci, H, W, logW16s[l]); break;
        case 1: conv_wmma_kernel<9, true><<<grid, threads, ldsB, stream>>>(
                    lin, wpk[l], cb[l], convbf, Ci, H, W, logW16s[l]); break;
        case 2: conv_wmma_kernel<18, false><<<grid, threads, ldsB, stream>>>(
                    lin, wpk[l], cb[l], convbf, Ci, H, W, logW16s[l]); break;
        case 3: conv_wmma_kernel<27, false><<<grid, threads, ldsB, stream>>>(
                    lin, wpk[l], cb[l], convbf, Ci, H, W, logW16s[l]); break;
      }
      zero_kernel<<<1, 64, 0, stream>>>(stats, 16);
      int chPerGroup = Co / 8;
      int HW = H * W;
      gn_stats_kernel<<<8 * 32, 256, 0, stream>>>(convbf, stats, chPerGroup, HW, 32);
      int totP = Co * (H / 2) * (W / 2);
      float invN = 1.f / ((float)chPerGroup * HW);
      gn_pool_kernel<<<(totP + 255) / 256, 256, 0, stream>>>(
          convbf, stats, gng[l], gnb[l], lpool[l], Co, H, W, chPerGroup, invN);
      lin = lpool[l];
    }
    avgpool4_kernel<<<32, 256, 0, stream>>>(poolB, feats + (size_t)b * 8192);
  }

  // FC head: pack feats, WMMA fc1, then fc2 + heads.
  pack_feats_kernel<<<(131072 + 255) / 256, 256, 0, stream>>>(feats, featpk);
  fc1_wmma_kernel<<<1, 256, 0, stream>>>(featpk, w1pk, b1, h1);
  fc2_heads_kernel<<<16, 64, 0, stream>>>(h1, w2, b2, ksw, ksb, rsw, rsb,
                                          x0w, x0b, y0w, y0b, scalars);

  // NFW fields -> output (16,3,512,512).
  {
    size_t tot = (size_t)16 * 512 * 512;
    nfw_kernel<<<(int)((tot + 255) / 256), 256, 0, stream>>>(scalars, out, 16, 512, 512);
  }
}
